// PerformerAttention_68118181315184
// MI455X (gfx1250) — matrix operand order repaired and checked by reading, not compiled, not run on hardware
//
#include <hip/hip_runtime.h>
#include <hip/hip_bf16.h>

// ---------------------------------------------------------------------------
// Performer attention for MI455X (gfx1250, wave32, WMMA).
// Heavy math: v_wmma_f32_16x16x32_bf16. The block-shared B tile is staged
// through LDS with global_load_async_to_lds_b128 (ASYNCcnt) in a 2x-unrolled,
// branchless double-buffered pipeline; A fragments are register
// double-buffered global loads. Wave tile 32(M) x 64(N): each LDS B fragment
// feeds two WMMAs.
// ---------------------------------------------------------------------------

typedef unsigned short u16;
typedef long long ll;
typedef __attribute__((ext_vector_type(16))) __bf16 v16bf;
typedef __attribute__((ext_vector_type(8)))  float  v8f;
typedef __attribute__((ext_vector_type(4)))  unsigned int u32x4;

union Frag { v16bf v; u32x4 q[2]; };

static constexpr int cB  = 4;
static constexpr int cS  = 4096;
static constexpr int cD  = 1024;
static constexpr int cH  = 16;
static constexpr int cF  = 256;
static constexpr int cHD = 64;
static constexpr int cM  = cB * cS;     // 16384 rows of x

// LDS B-tile: 64 columns x 32 K-halves, padded stride to spread banks.
static constexpr int BSTRIDE = 40;      // halves per column (32 + 8 pad)

__device__ __forceinline__ u16 f2bf(float f) {
    union { float f; unsigned u; } a; a.f = f;
    unsigned r = a.u + 0x7FFFu + ((a.u >> 16) & 1u);   // round-to-nearest-even
    return (u16)(r >> 16);
}
__device__ __forceinline__ float bf2f(u16 h) {
    union { unsigned u; float f; } a; a.u = ((unsigned)h) << 16;
    return a.f;
}
__device__ __forceinline__ float softplus_f(float x) {
    return (x > 20.0f) ? x : __logf(1.0f + __expf(x));
}

__device__ __forceinline__ void wait_async0() {
#if __has_builtin(__builtin_amdgcn_s_wait_asynccnt)
    __builtin_amdgcn_s_wait_asynccnt(0);
#else
    asm volatile("s_wait_asynccnt 0x0" ::: "memory");
#endif
}

// Async copy of 16 bytes global -> LDS (VDST = LDS byte address, VADDR = vaddr)
__device__ __forceinline__ void async_g2l_b128(unsigned lds_byte_addr,
                                               const u16* gaddr) {
    asm volatile("global_load_async_to_lds_b128 %0, %1, off"
                 :: "v"(lds_byte_addr), "v"(gaddr) : "memory");
}

// ---------------------------------------------------------------------------
// fp32 -> bf16 conversion (grid-stride)
// ---------------------------------------------------------------------------
__global__ void conv_f32_bf16(const float* __restrict__ src,
                              u16* __restrict__ dst, int n) {
    int i = blockIdx.x * blockDim.x + threadIdx.x;
    int stride = gridDim.x * blockDim.x;
    for (; i < n; i += stride) dst[i] = f2bf(src[i]);
}

// rf [H, HD, F] fp32 -> rfT [H, F, HD] bf16 (feature GEMM B contiguous along K)
__global__ void conv_rf_T(const float* __restrict__ src, u16* __restrict__ dst) {
    int i = blockIdx.x * blockDim.x + threadIdx.x;
    if (i >= cH * cHD * cF) return;
    int h = i / (cHD * cF);
    int r = i % (cHD * cF);
    int d = r / cF;
    int f = r % cF;
    dst[(h * cF + f) * cHD + d] = f2bf(src[i]);
}

// ---------------------------------------------------------------------------
// bf16 WMMA GEMM:  C[z] = A[z] (MxK row-major, lda) x B[z] (KxN column-major:
// column n starts at B + n*ldb, contiguous along K).
// Block = 256 threads = 8 waves; block tile 256(M) x 64(N); wave tile 32x64.
// STORE_MODE: 0 = C[m*ldc+n]; 1 = C[n*ldc+m];
//             2 = head scatter  [B,H,S,HD] from (m=b*S+s, n=h*64+d)
//             3 = head scatter T[B,H,HD,S]
// ACT: 0 none, 1 softplus.  OUT_BF16 / HAS_BIAS: output dtype + bias add.
// Requires M % 256 == 0, N % 64 == 0, K % 64 == 0 (true for all uses), so all
// waves run the whole pipeline and EXEC is all-1s at every WMMA/barrier.
// ---------------------------------------------------------------------------
template<int STORE_MODE, int ACT, bool OUT_BF16, bool HAS_BIAS>
__global__ void gemm_bf16_wmma(const u16* __restrict__ A, int lda, ll strideAz,
                               const u16* __restrict__ Bm, int ldb, ll strideBz,
                               int bModH,
                               const float* __restrict__ bias,
                               float* __restrict__ Cf, u16* __restrict__ Cb,
                               int ldc, ll strideCz,
                               int M, int N, int K, int Sp, int Hp) {
    __shared__ u16 sB[2][64 * BSTRIDE];

    const int tid  = threadIdx.x;
    const int wave = tid >> 5;
    const int lane = tid & 31;
    const int z    = blockIdx.z;
    const int mBase = blockIdx.y * 256 + wave * 32;
    const int nBase = blockIdx.x * 64;

    const u16* Ab = A  + (ll)z * strideAz;
    const u16* Bb = Bm + (ll)(bModH ? (z % Hp) : z) * strideBz;

    const int kA = (lane >> 4) << 3;             // A: 0 or 8   (16x32 bf16 layout)
    const int kB = (lane >> 4) << 3;             // B: 0 or 8   (32x16 bf16 layout)

    const u16* aRow0 = Ab + (ll)(mBase + (lane & 15)) * lda + kA;        // M 0..15
    const u16* aRow1 = Ab + (ll)(mBase + 16 + (lane & 15)) * lda + kA;   // M 16..31

    // LDS base byte address (flat shared pointer low 32 bits == LDS offset)
    const unsigned ldsBase = (unsigned)(uintptr_t)(&sB[0][0]);

    // cooperative async stage of one 64x32 B tile: thread t copies 16 bytes
    const int stC  = tid >> 2;                   // column 0..63
    const int stCh = tid & 3;                    // 8-half chunk 0..3
    auto stage = [&](int buf, int k) {
        const u16* g = Bb + (ll)(nBase + stC) * ldb + k + stCh * 8;
        unsigned l = ldsBase +
                     ((unsigned)buf * (64 * BSTRIDE) + stC * BSTRIDE + stCh * 8) * 2u;
        async_g2l_b128(l, g);
    };
    auto loadA = [&](Frag& f0, Frag& f1, int step) {
        const u16* p0 = aRow0 + (ll)step * 32;
        const u16* p1 = aRow1 + (ll)step * 32;
        f0.q[0] = *(const u32x4*)(p0); f0.q[1] = *(const u32x4*)(p0 + 16);
        f1.q[0] = *(const u32x4*)(p1); f1.q[1] = *(const u32x4*)(p1 + 16);
    };

    v8f acc[8];
    {
        v8f zz = {0, 0, 0, 0, 0, 0, 0, 0};
#pragma unroll
        for (int t = 0; t < 8; ++t) acc[t] = zz;
    }

    // acc[t] = M-rows 0..15, acc[4+t] = M-rows 16..31 of n-tile t.
    auto computeStep = [&](const u16* sbuf, const Frag& a0, const Frag& a1) {
        Frag b[4];
#pragma unroll
        for (int t = 0; t < 4; ++t) {            // issue all 8 ds_loads first
            const int cl = t * 16 + (lane & 15);
            b[t].q[0] = *(const u32x4*)(sbuf + cl * BSTRIDE + kB);
            b[t].q[1] = *(const u32x4*)(sbuf + cl * BSTRIDE + kB + 16);
        }
#pragma unroll
        for (int t = 0; t < 4; ++t) {
            acc[t]     = __builtin_amdgcn_wmma_f32_16x16x32_bf16(
                             false, a0.v, false, b[t].v, (short)0, acc[t],
                             false, false);
            acc[4 + t] = __builtin_amdgcn_wmma_f32_16x16x32_bf16(
                             false, a1.v, false, b[t].v, (short)0, acc[4 + t],
                             false, false);
        }
    };

    const int nSteps = K / 32;                   // always even, >= 2
    Frag aE0, aE1, aO0, aO1;

    stage(0, 0);
    loadA(aE0, aE1, 0);

    int i = 0;
    for (; i + 2 < nSteps; i += 2) {
        wait_async0();                           // buf0 (step i) landed
        __syncthreads();
        stage(1, (i + 1) * 32);
        loadA(aO0, aO1, i + 1);
        computeStep(&sB[0][0], aE0, aE1);

        wait_async0();                           // buf1 (step i+1) landed
        __syncthreads();
        stage(0, (i + 2) * 32);
        loadA(aE0, aE1, i + 2);
        computeStep(&sB[1][0], aO0, aO1);
    }
    // final pair (i == nSteps - 2)
    wait_async0();
    __syncthreads();
    stage(1, (i + 1) * 32);
    loadA(aO0, aO1, i + 1);
    computeStep(&sB[0][0], aE0, aE1);

    wait_async0();
    __syncthreads();
    computeStep(&sB[1][0], aO0, aO1);

    // ---- epilogue ---------------------------------------------------------
#pragma unroll
    for (int t = 0; t < 4; ++t) {
        const int n = nBase + t * 16 + (lane & 15);
        const float bv = HAS_BIAS ? bias[n] : 0.0f;
#pragma unroll
        for (int mh = 0; mh < 2; ++mh) {
#pragma unroll
            for (int r = 0; r < 8; ++r) {
                const int m = mBase + mh * 16 + r + ((lane >> 4) << 3);
                float val = acc[mh * 4 + t][r] + bv;
                if (ACT == 1) val = softplus_f(val);
                ll idx;
                if (STORE_MODE == 0) {
                    idx = (ll)z * strideCz + (ll)m * ldc + n;
                } else if (STORE_MODE == 1) {
                    idx = (ll)z * strideCz + (ll)n * ldc + m;
                } else {
                    const int b = m / Sp, s = m % Sp, h = n >> 6, d = n & 63;
                    if (STORE_MODE == 2)
                        idx = ((((ll)b * Hp + h) * Sp + s) << 6) + d;     // [B,H,S,HD]
                    else
                        idx = ((((ll)b * Hp + h) << 6) + d) * (ll)Sp + s; // [B,H,HD,S]
                }
                if (OUT_BF16) Cb[idx] = f2bf(val);
                else          Cf[idx] = val;
            }
        }
    }
}

// ---------------------------------------------------------------------------
// k_sum[z,f] = sum_s k'[z,f,s]  (k' stored transposed [B*H, F, S], bf16)
// ---------------------------------------------------------------------------
__global__ void ksum_kernel(const u16* __restrict__ kpT, float* __restrict__ ksum) {
    const int gw   = (blockIdx.x * blockDim.x + threadIdx.x) >> 5; // [0, B*H*F)
    const int lane = threadIdx.x & 31;
    const u16* row = kpT + (ll)gw * cS;
    float part = 0.0f;
#pragma unroll
    for (int i = 0; i < cS / (32 * 8); ++i) {         // 16 iterations
        const u32x4 c = *(const u32x4*)(row + (i * 32 + lane) * 8);
#pragma unroll
        for (int j = 0; j < 4; ++j) {
            unsigned u = c[j];
            part += bf2f((u16)(u & 0xFFFFu)) + bf2f((u16)(u >> 16));
        }
    }
    for (int off = 16; off; off >>= 1) part += __shfl_xor(part, off, 32);
    if (lane == 0) ksum[gw] = part;
}

// ---------------------------------------------------------------------------
// denom = q'[s,:] . k_sum + 1e-8 ;  ctx[b,s,h*64+d] = qkv[z,s,d] / denom (bf16)
// ---------------------------------------------------------------------------
__global__ void normalize_kernel(const u16* __restrict__ qp,
                                 const float* __restrict__ ksum,
                                 const float* __restrict__ qkv,
                                 u16* __restrict__ ctx) {
    const int gw   = (blockIdx.x * blockDim.x + threadIdx.x) >> 5; // [0, B*H*S)
    const int lane = threadIdx.x & 31;
    const int z = gw / cS;
    const int s = gw % cS;

    const u16*   qrow = qp + ((ll)z * cS + s) * cF;
    const float* ks   = ksum + z * cF;

    float part = 0.0f;
    const u32x4 c = *(const u32x4*)(qrow + lane * 8);
#pragma unroll
    for (int j = 0; j < 4; ++j) {
        unsigned u = c[j];
        part += bf2f((u16)(u & 0xFFFFu)) * ks[lane * 8 + 2 * j]
              + bf2f((u16)(u >> 16))     * ks[lane * 8 + 2 * j + 1];
    }
    for (int off = 16; off; off >>= 1) part += __shfl_xor(part, off, 32);

    const float inv = 1.0f / (part + 1e-8f);
    const int b = z / cH, h = z % cH;
    const float* qr = qkv + (((ll)z * cS + s) << 6);
    u16* crow = ctx + (((ll)b * cS + s) << 10) + (h << 6);
#pragma unroll
    for (int j = 0; j < 2; ++j) {
        const int d = lane + 32 * j;
        crow[d] = f2bf(qr[d] * inv);
    }
}

// ---------------------------------------------------------------------------
// Host orchestration
// ---------------------------------------------------------------------------
extern "C" void kernel_launch(void* const* d_in, const int* in_sizes, int n_in,
                              void* d_out, int out_size, void* d_ws, size_t ws_size,
                              hipStream_t stream) {
    (void)in_sizes; (void)n_in; (void)out_size; (void)ws_size;

    const float* x  = (const float*)d_in[0];
    const float* Wq = (const float*)d_in[1];
    const float* bq = (const float*)d_in[2];
    const float* Wk = (const float*)d_in[3];
    const float* bk = (const float*)d_in[4];
    const float* Wv = (const float*)d_in[5];
    const float* bv = (const float*)d_in[6];
    const float* Wo = (const float*)d_in[7];
    const float* bo = (const float*)d_in[8];
    const float* rf = (const float*)d_in[9];
    float* out = (float*)d_out;

    char* w = (char*)d_ws;
    size_t off = 0;
    auto alloc = [&](size_t bytes) -> void* {
        void* p = w + off;
        off += (bytes + 255) & ~(size_t)255;
        return p;
    };

    u16*   xb   = (u16*)  alloc((size_t)cM * cD * 2);            // x bf16
    u16*   wqb  = (u16*)  alloc((size_t)cD * cD * 2);
    u16*   wkb  = (u16*)  alloc((size_t)cD * cD * 2);
    u16*   wvb  = (u16*)  alloc((size_t)cD * cD * 2);
    u16*   wob  = (u16*)  alloc((size_t)cD * cD * 2);
    u16*   rfT  = (u16*)  alloc((size_t)cH * cF * cHD * 2);      // [H,F,HD]
    u16*   qh   = (u16*)  alloc((size_t)cB * cH * cS * cHD * 2); // [B,H,S,HD]
    u16*   kh   = (u16*)  alloc((size_t)cB * cH * cS * cHD * 2); // [B,H,S,HD]
    u16*   vT   = (u16*)  alloc((size_t)cB * cH * cHD * cS * 2); // [B,H,HD,S]
    u16*   qp   = (u16*)  alloc((size_t)cB * cH * cS * cF * 2);  // [B,H,S,F]
    u16*   kpT  = (u16*)  alloc((size_t)cB * cH * cF * cS * 2);  // [B,H,F,S]
    u16*   kvT  = (u16*)  alloc((size_t)cB * cH * cHD * cF * 2); // [B,H,HD,F]
    float* ksum = (float*)alloc((size_t)cB * cH * cF * 4);       // [B,H,F]
    float* qkvb = (float*)alloc((size_t)cB * cH * cS * cHD * 4); // [B,H,S,HD]
    u16*   ctx  = (u16*)  alloc((size_t)cM * cD * 2);            // [B,S,D]

    // --- conversions -------------------------------------------------------
    conv_f32_bf16<<<4096, 256, 0, stream>>>(x, xb, cM * cD);
    conv_f32_bf16<<<1024, 256, 0, stream>>>(Wq, wqb, cD * cD);
    conv_f32_bf16<<<1024, 256, 0, stream>>>(Wk, wkb, cD * cD);
    conv_f32_bf16<<<1024, 256, 0, stream>>>(Wv, wvb, cD * cD);
    conv_f32_bf16<<<1024, 256, 0, stream>>>(Wo, wob, cD * cD);
    conv_rf_T<<<(cH * cHD * cF + 255) / 256, 256, 0, stream>>>(rf, rfT);

    // --- Q/K/V projections: y = x @ W.T + b --------------------------------
    {
        dim3 g(cD / 64, cM / 256, 1);
        gemm_bf16_wmma<2, 0, true, true><<<g, 256, 0, stream>>>(
            xb, cD, 0, wqb, cD, 0, 0, bq, nullptr, qh, 0, 0,
            cM, cD, cD, cS, cH);
        gemm_bf16_wmma<2, 0, true, true><<<g, 256, 0, stream>>>(
            xb, cD, 0, wkb, cD, 0, 0, bk, nullptr, kh, 0, 0,
            cM, cD, cD, cS, cH);
        gemm_bf16_wmma<3, 0, true, true><<<g, 256, 0, stream>>>(
            xb, cD, 0, wvb, cD, 0, 0, bv, nullptr, vT, 0, 0,
            cM, cD, cD, cS, cH);
    }

    // --- feature maps: q' = softplus(q @ rf[h]), k' likewise (k' stored T) --
    {
        dim3 g(cF / 64, cS / 256, cB * cH);
        gemm_bf16_wmma<0, 1, true, false><<<g, 256, 0, stream>>>(
            qh, cHD, (ll)cS * cHD, rfT, cHD, (ll)cF * cHD, 1, nullptr,
            nullptr, qp, cF, (ll)cS * cF, cS, cF, cHD, cS, cH);
        gemm_bf16_wmma<1, 1, true, false><<<g, 256, 0, stream>>>(
            kh, cHD, (ll)cS * cHD, rfT, cHD, (ll)cF * cHD, 1, nullptr,
            nullptr, kpT, cS, (ll)cF * cS, cS, cF, cHD, cS, cH);
    }

    // --- k_sum -------------------------------------------------------------
    ksum_kernel<<<(cB * cH * cF * 32) / 256, 256, 0, stream>>>(kpT, ksum);

    // --- kv = k'^T @ v  -> stored transposed [B,H,HD,F] --------------------
    {
        dim3 g(cHD / 64, cF / 256, cB * cH);
        gemm_bf16_wmma<1, 0, true, false><<<g, 256, 0, stream>>>(
            kpT, cS, (ll)cF * cS, vT, cS, (ll)cHD * cS, 0, nullptr,
            nullptr, kvT, cF, (ll)cHD * cF, cF, cHD, cS, cS, cH);
    }

    // --- qkv = q' @ kv  (fp32 out) -----------------------------------------
    {
        dim3 g(cHD / 64, cS / 256, cB * cH);
        gemm_bf16_wmma<0, 0, false, false><<<g, 256, 0, stream>>>(
            qp, cF, (ll)cS * cF, kvT, cF, (ll)cHD * cF, 0, nullptr,
            qkvb, nullptr, cHD, (ll)cS * cHD, cS, cHD, cF, cS, cH);
    }

    // --- normalize + merge heads into ctx [B,S,D] bf16 ---------------------
    normalize_kernel<<<(cB * cH * cS * 32) / 256, 256, 0, stream>>>(
        qp, ksum, qkvb, ctx);

    // --- output projection: out = ctx @ Wo.T + bo (fp32 to d_out) ----------
    {
        dim3 g(cD / 64, cM / 256, 1);
        gemm_bf16_wmma<0, 0, false, true><<<g, 256, 0, stream>>>(
            ctx, cD, 0, wob, cD, 0, 0, bo, out, nullptr, cD, 0,
            cM, cD, cD, cS, cH);
    }
}
